// PolyAlexNet_6476810682808
// MI455X (gfx1250) — matrix operand order repaired and checked by reading, not compiled, not run on hardware
//
#include <hip/hip_runtime.h>
#include <hip/hip_bf16.h>

// ---------------------------------------------------------------------------
// PolyAlexNet forward on gfx1250 — panel-packed f16 WMMA GEMM with register
// blocking. Pack kernels pre-swizzle A (weights / FC activations) and B
// (im2col / dense FC weights) into a per-lane fragment layout (contiguous k = 16*half + e) for
// v_wmma_f32_16x16x32_f16 (ISA 7.12.2), zero-padded in K/M/N. The GEMM wave
// computes a 64(M) x 32(N) macro-tile: per k-step it loads 4 A-frags + 2
// B-frags (contiguous 32B each -> global_load_b128 pairs) and issues 8 WMMAs,
// giving ~21 FLOP/byte of L2 fragment traffic instead of 8.
// Epilogue fuses the PolyKervNets activation: (acc + bias + 1)^2.
// ---------------------------------------------------------------------------

typedef __attribute__((ext_vector_type(16))) _Float16 v16h;
typedef __attribute__((ext_vector_type(8)))  float    v8f;

#define MBLK 4   // M tiles per wave (64 rows)
#define NBLK 2   // N tiles per wave (32 cols)

// ---------------------------------------------------------------------------
// Panel order (both panels): dst[((tile*Ksteps + ks)*32 + lane)*16 + e]
// A-fragment (16-bit A 16x32): m = tile*16 + (lane&15)
//   kl = (lane>>4)*16 + e
// B-fragment (16-bit B 32x16): n = tile*16 + (lane&15); kl = (lane>>4)*16 + e
// ---------------------------------------------------------------------------

__global__ void pack_a_f16(const float* __restrict__ src, _Float16* __restrict__ dst,
                           int Mrows, int K, int Ksteps, long long total)
{
    const long long i = (long long)blockIdx.x * blockDim.x + threadIdx.x;
    if (i >= total) return;

    const int e    = (int)(i & 15);
    const int lane = (int)((i >> 4) & 31);
    const long long q = i >> 9;                 // tile*Ksteps + ks
    const int ks   = (int)(q % Ksteps);
    const int mt   = (int)(q / Ksteps);

    const int half = lane >> 4, l16 = lane & 15;
    const int m = mt * 16 + l16;
    const int j = e >> 1, t = e & 1;
    const int kl = half * 16 + e;
    const int k = ks * 32 + kl;

    float v = 0.0f;
    if (m < Mrows && k < K) v = src[(size_t)m * K + k];
    dst[i] = (_Float16)v;
}

__global__ void pack_b_im2col(const float* __restrict__ x, _Float16* __restrict__ dst,
                              int n0, int Ntotal, int K, int Ksteps,
                              int Cin, int H, int W, int KH, int KW,
                              int stride, int pad, int OH, int OW, long long total)
{
    const long long i = (long long)blockIdx.x * blockDim.x + threadIdx.x;
    if (i >= total) return;

    const int e    = (int)(i & 15);
    const int lane = (int)((i >> 4) & 31);
    const long long q = i >> 9;
    const int ks   = (int)(q % Ksteps);
    const int nt   = (int)(q / Ksteps);

    const int half = lane >> 4, l16 = lane & 15;
    const int n  = n0 + nt * 16 + l16;
    const int kl = half * 16 + e;
    const int k  = ks * 32 + kl;

    float v = 0.0f;
    if (n < Ntotal && k < K) {
        const int ohw = OH * OW;
        const int b  = n / ohw;
        const int r  = n - b * ohw;
        const int oh = r / OW;
        const int ow = r - oh * OW;

        const int khw = KH * KW;
        const int ci = k / khw;
        const int rr = k - ci * khw;
        const int kh = rr / KW;
        const int kw = rr - kh * KW;

        const int ih = oh * stride - pad + kh;
        const int iw = ow * stride - pad + kw;
        if (ih >= 0 && ih < H && iw >= 0 && iw < W)
            v = x[(((size_t)b * Cin + ci) * H + ih) * W + iw];
    }
    dst[i] = (_Float16)v;
}

__global__ void pack_b_dense(const float* __restrict__ src, _Float16* __restrict__ dst,
                             int n0, int N, int K, int Ksteps, long long total)
{
    const long long i = (long long)blockIdx.x * blockDim.x + threadIdx.x;
    if (i >= total) return;

    const int e    = (int)(i & 15);
    const int lane = (int)((i >> 4) & 31);
    const long long q = i >> 9;
    const int ks   = (int)(q % Ksteps);
    const int nt   = (int)(q / Ksteps);

    const int half = lane >> 4, l16 = lane & 15;
    const int n  = n0 + nt * 16 + l16;
    const int kl = half * 16 + e;
    const int k  = ks * 32 + kl;

    float v = 0.0f;
    if (n < N && k < K) v = src[(size_t)k * N + n];
    dst[i] = (_Float16)v;
}

// ---------------------------------------------------------------------------
// Register-blocked GEMM mainloop: 4 A-frags + 2 B-frags -> 8 WMMAs / k-step.
// ---------------------------------------------------------------------------
__device__ __forceinline__ void gemm_mainloop_4x2(
    const _Float16* __restrict__ Ap, const _Float16* __restrict__ Bp,
    int mblk, int nblk, int Ksteps, int lane, v8f acc[MBLK][NBLK])
{
    const size_t tileStride = (size_t)Ksteps * 512;   // f16 elems per tile-panel

    const _Float16* a0 = Ap + (size_t)(mblk * MBLK + 0) * tileStride + (size_t)lane * 16;
    const _Float16* a1 = a0 + tileStride;
    const _Float16* a2 = a1 + tileStride;
    const _Float16* a3 = a2 + tileStride;
    const _Float16* b0 = Bp + (size_t)(nblk * NBLK + 0) * tileStride + (size_t)lane * 16;
    const _Float16* b1 = b0 + tileStride;

    for (int ks = 0; ks < Ksteps; ++ks) {
        const v16h fa0 = *(const v16h*)a0;
        const v16h fa1 = *(const v16h*)a1;
        const v16h fa2 = *(const v16h*)a2;
        const v16h fa3 = *(const v16h*)a3;
        const v16h fb0 = *(const v16h*)b0;
        const v16h fb1 = *(const v16h*)b1;
        a0 += 512; a1 += 512; a2 += 512; a3 += 512;
        b0 += 512; b1 += 512;
        __builtin_prefetch(a0, 0, 1);               // global_prefetch_b8
        __builtin_prefetch(b0, 0, 1);

        acc[0][0] = __builtin_amdgcn_wmma_f32_16x16x32_f16(false, fa0, false, fb0, (short)0, acc[0][0], false, false);
        acc[1][0] = __builtin_amdgcn_wmma_f32_16x16x32_f16(false, fa1, false, fb0, (short)0, acc[1][0], false, false);
        acc[2][0] = __builtin_amdgcn_wmma_f32_16x16x32_f16(false, fa2, false, fb0, (short)0, acc[2][0], false, false);
        acc[3][0] = __builtin_amdgcn_wmma_f32_16x16x32_f16(false, fa3, false, fb0, (short)0, acc[3][0], false, false);
        acc[0][1] = __builtin_amdgcn_wmma_f32_16x16x32_f16(false, fa0, false, fb1, (short)0, acc[0][1], false, false);
        acc[1][1] = __builtin_amdgcn_wmma_f32_16x16x32_f16(false, fa1, false, fb1, (short)0, acc[1][1], false, false);
        acc[2][1] = __builtin_amdgcn_wmma_f32_16x16x32_f16(false, fa2, false, fb1, (short)0, acc[2][1], false, false);
        acc[3][1] = __builtin_amdgcn_wmma_f32_16x16x32_f16(false, fa3, false, fb1, (short)0, acc[3][1], false, false);
    }
}

// Conv epilogue: y = (acc + bias[m] + 1)^2, scatter NCHW.
__global__ void __launch_bounds__(32)
gemm_conv_wmma(const _Float16* __restrict__ Ap, const _Float16* __restrict__ Bp,
               const float* __restrict__ bias, float* __restrict__ y,
               int Ksteps, int n0, int OC, int Ntotal, int OH, int OW)
{
    const int lane = threadIdx.x & 31;
    const int half = lane >> 4, l16 = lane & 15;

    v8f acc[MBLK][NBLK] = {};
    gemm_mainloop_4x2(Ap, Bp, blockIdx.x, blockIdx.y, Ksteps, lane, acc);

    const int ohw = OH * OW;
#pragma unroll
    for (int ni = 0; ni < NBLK; ++ni) {
        const int n = n0 + ((int)blockIdx.y * NBLK + ni) * 16 + l16;
        if (n >= Ntotal) continue;
        const int b  = n / ohw;
        const int r  = n - b * ohw;
        const int oh = r / OW;
        const int ow = r - oh * OW;
#pragma unroll
        for (int mi = 0; mi < MBLK; ++mi) {
#pragma unroll
            for (int v = 0; v < 8; ++v) {
                const int m = ((int)blockIdx.x * MBLK + mi) * 16 + v + half * 8;
                if (m < OC) {
                    float t = acc[mi][ni][v] + bias[m] + 1.0f;  // BALANCE = 1
                    t = t * t;                                   // POWER   = 2
                    y[(((size_t)b * OC + m) * OH + oh) * OW + ow] = t;
                }
            }
        }
    }
}

// FC epilogue: C[m,n] = acc + bias[n], row-major.
__global__ void __launch_bounds__(32)
gemm_fc_wmma(const _Float16* __restrict__ Ap, const _Float16* __restrict__ Bp,
             const float* __restrict__ bias, float* __restrict__ C,
             int Ksteps, int n0, int M, int N)
{
    const int lane = threadIdx.x & 31;
    const int half = lane >> 4, l16 = lane & 15;

    v8f acc[MBLK][NBLK] = {};
    gemm_mainloop_4x2(Ap, Bp, blockIdx.x, blockIdx.y, Ksteps, lane, acc);

#pragma unroll
    for (int ni = 0; ni < NBLK; ++ni) {
        const int n = n0 + ((int)blockIdx.y * NBLK + ni) * 16 + l16;
        if (n >= N) continue;
        const float bn = bias[n];
#pragma unroll
        for (int mi = 0; mi < MBLK; ++mi) {
#pragma unroll
            for (int v = 0; v < 8; ++v) {
                const int m = ((int)blockIdx.x * MBLK + mi) * 16 + v + half * 8;
                if (m < M)
                    C[(size_t)m * N + n] = acc[mi][ni][v] + bn;
            }
        }
    }
}

// ---------------------------------------------------------------------------
// avgpool 3x3 stride 2 (VALID) on NCHW — pure bandwidth.
// ---------------------------------------------------------------------------
__global__ void avgpool3s2_k(const float* __restrict__ x, float* __restrict__ y,
                             int B, int C, int H, int W, int OH, int OW)
{
    const size_t i     = (size_t)blockIdx.x * blockDim.x + threadIdx.x;
    const size_t total = (size_t)B * C * OH * OW;
    if (i >= total) return;

    const int ow = (int)(i % OW);
    size_t r = i / OW;
    const int oh = (int)(r % OH);
    r /= OH;
    const int c = (int)(r % C);
    const int b = (int)(r / C);

    const float* p = x + (((size_t)b * C + c) * H + (size_t)oh * 2) * W + (size_t)ow * 2;
    float s = 0.0f;
#pragma unroll
    for (int hh = 0; hh < 3; ++hh)
#pragma unroll
        for (int ww = 0; ww < 3; ++ww)
            s += p[(size_t)hh * W + ww];
    y[i] = s * (1.0f / 9.0f);
}

// ---------------------------------------------------------------------------
// Host-side orchestration
// ---------------------------------------------------------------------------
static inline unsigned cdiv_u(long long a, long long b) {
    return (unsigned)((a + b - 1) / b);
}
static inline int round_up(int v, int m) { return ((v + m - 1) / m) * m; }

static const size_t PB_CAP = 16ull * 1024 * 1024;   // f16 elems (32 MB) B-panel
static const size_t PA_CAP = 2ull  * 1024 * 1024;   // f16 elems (4 MB)  A-panel

static void run_conv(const float* x, const float* w, const float* bias, float* y,
                     int B, int Cin, int H, int W, int OC, int KH, int KW,
                     int stride, int pad, int OH, int OW,
                     _Float16* PA, _Float16* PB, hipStream_t stream)
{
    const int K         = Cin * KH * KW;
    const int Ksteps    = (K + 31) / 32;
    const int N         = B * OH * OW;
    const int Ntiles    = (N + 15) / 16;
    const int MtilesPad = round_up((OC + 15) / 16, MBLK);

    // Pack weights into A panel (once per layer, zero-padded to MBLK tiles).
    {
        const long long tot = (long long)MtilesPad * Ksteps * 512;
        pack_a_f16<<<cdiv_u(tot, 256), 256, 0, stream>>>(w, PA, OC, K, Ksteps, tot);
    }

    int chunkTiles = (int)(PB_CAP / ((size_t)Ksteps * 512));
    chunkTiles &= ~(NBLK - 1);                        // keep even
    if (chunkTiles < NBLK) chunkTiles = NBLK;
    if (chunkTiles > round_up(Ntiles, NBLK)) chunkTiles = round_up(Ntiles, NBLK);

    for (int t0 = 0; t0 < Ntiles; t0 += chunkTiles) {
        const int nt    = (t0 + chunkTiles <= Ntiles) ? chunkTiles : (Ntiles - t0);
        const int ntPad = round_up(nt, NBLK);
        const int n0    = t0 * 16;
        const long long tot = (long long)ntPad * Ksteps * 512;
        pack_b_im2col<<<cdiv_u(tot, 256), 256, 0, stream>>>(
            x, PB, n0, N, K, Ksteps, Cin, H, W, KH, KW, stride, pad, OH, OW, tot);
        dim3 g((unsigned)(MtilesPad / MBLK), (unsigned)(ntPad / NBLK));
        gemm_conv_wmma<<<g, 32, 0, stream>>>(PA, PB, bias, y,
                                             Ksteps, n0, OC, N, OH, OW);
    }
}

static void run_fc(const float* A, const float* Wm, const float* bias, float* C,
                   int M, int N, int K, _Float16* PA, _Float16* PB, hipStream_t stream)
{
    const int Ksteps    = (K + 31) / 32;
    const int Ntiles    = (N + 15) / 16;
    const int MtilesPad = round_up((M + 15) / 16, MBLK);

    {
        const long long tot = (long long)MtilesPad * Ksteps * 512;
        pack_a_f16<<<cdiv_u(tot, 256), 256, 0, stream>>>(A, PA, M, K, Ksteps, tot);
    }

    int chunkTiles = (int)(PB_CAP / ((size_t)Ksteps * 512));
    chunkTiles &= ~(NBLK - 1);
    if (chunkTiles < NBLK) chunkTiles = NBLK;
    if (chunkTiles > round_up(Ntiles, NBLK)) chunkTiles = round_up(Ntiles, NBLK);

    for (int t0 = 0; t0 < Ntiles; t0 += chunkTiles) {
        const int nt    = (t0 + chunkTiles <= Ntiles) ? chunkTiles : (Ntiles - t0);
        const int ntPad = round_up(nt, NBLK);
        const int n0    = t0 * 16;
        const long long tot = (long long)ntPad * Ksteps * 512;
        pack_b_dense<<<cdiv_u(tot, 256), 256, 0, stream>>>(
            Wm, PB, n0, N, K, Ksteps, tot);
        dim3 g((unsigned)(MtilesPad / MBLK), (unsigned)(ntPad / NBLK));
        gemm_fc_wmma<<<g, 32, 0, stream>>>(PA, PB, bias, C, Ksteps, n0, M, N);
    }
}

extern "C" void kernel_launch(void* const* d_in, const int* in_sizes, int n_in,
                              void* d_out, int out_size, void* d_ws, size_t ws_size,
                              hipStream_t stream) {
    (void)in_sizes; (void)n_in; (void)out_size; (void)ws_size;

    const float* x   = (const float*)d_in[0];
    const float* w1  = (const float*)d_in[1];
    const float* b1  = (const float*)d_in[2];
    const float* w2  = (const float*)d_in[3];
    const float* b2  = (const float*)d_in[4];
    const float* w3  = (const float*)d_in[5];
    const float* b3  = (const float*)d_in[6];
    const float* w4  = (const float*)d_in[7];
    const float* b4  = (const float*)d_in[8];
    const float* w5  = (const float*)d_in[9];
    const float* b5  = (const float*)d_in[10];
    const float* fw1 = (const float*)d_in[11];
    const float* fb1 = (const float*)d_in[12];
    const float* fw2 = (const float*)d_in[13];
    const float* fb2 = (const float*)d_in[14];
    const float* fw3 = (const float*)d_in[15];
    const float* fb3 = (const float*)d_in[16];
    float* out = (float*)d_out;

    // Workspace layout:
    //   R0, R1 : activation ping-pong, 18M floats each (max live = conv1 out)
    //   PA     : A-fragment panel (4 MB)
    //   PB     : B-fragment panel (32 MB, chunked)
    const size_t REGION_F = 18ull * 1024 * 1024;
    float*     R0 = (float*)d_ws;
    float*     R1 = R0 + REGION_F;
    _Float16*  PA = (_Float16*)(R1 + REGION_F);
    _Float16*  PB = PA + PA_CAP;

    const int B = 64;

    // conv1: x[64,3,227,227] -> R0[64,96,55,55]   s4 p0
    run_conv(x, w1, b1, R0, B, 3, 227, 227, 96, 11, 11, 4, 0, 55, 55, PA, PB, stream);
    // pool1 -> R1[64,96,27,27]
    {
        const long long tot = (long long)B * 96 * 27 * 27;
        avgpool3s2_k<<<cdiv_u(tot, 256), 256, 0, stream>>>(R0, R1, B, 96, 55, 55, 27, 27);
    }
    // conv2: R1 -> R0[64,256,27,27]  s1 p2
    run_conv(R1, w2, b2, R0, B, 96, 27, 27, 256, 5, 5, 1, 2, 27, 27, PA, PB, stream);
    // pool2 -> R1[64,256,13,13]
    {
        const long long tot = (long long)B * 256 * 13 * 13;
        avgpool3s2_k<<<cdiv_u(tot, 256), 256, 0, stream>>>(R0, R1, B, 256, 27, 27, 13, 13);
    }
    // conv3: R1 -> R0[64,384,13,13]  s1 p1
    run_conv(R1, w3, b3, R0, B, 256, 13, 13, 384, 3, 3, 1, 1, 13, 13, PA, PB, stream);
    // conv4: R0 -> R1[64,384,13,13]
    run_conv(R0, w4, b4, R1, B, 384, 13, 13, 384, 3, 3, 1, 1, 13, 13, PA, PB, stream);
    // conv5: R1 -> R0[64,256,13,13]
    run_conv(R1, w5, b5, R0, B, 384, 13, 13, 256, 3, 3, 1, 1, 13, 13, PA, PB, stream);
    // pool3 -> R1[64,256,6,6] == [64,9216] flat (NCHW contiguous)
    {
        const long long tot = (long long)B * 256 * 6 * 6;
        avgpool3s2_k<<<cdiv_u(tot, 256), 256, 0, stream>>>(R0, R1, B, 256, 13, 13, 6, 6);
    }
    // fc1: R1[64,9216] @ fw1 + fb1 -> R0[64,4096]
    run_fc(R1, fw1, fb1, R0, B, 4096, 9216, PA, PB, stream);
    // fc2: R0 @ fw2 + fb2 -> R1[64,4096]
    run_fc(R0, fw2, fb2, R1, B, 4096, 4096, PA, PB, stream);
    // fc3: R1 @ fw3 + fb3 -> out[64,1000]
    run_fc(R1, fw3, fb3, out, B, 1000, 4096, PA, PB, stream);
}
